// EGNN_vel_40458591929146
// MI455X (gfx1250) — matrix operand order repaired and checked by reading, not compiled, not run on hardware
//
#include <hip/hip_runtime.h>
#include <stdint.h>

// ---------------------------------------------------------------------------
// EGNN_vel for MI455X (gfx1250): bf16 WMMA (16x16x32) GEMMs, f32 accumulate.
// One wave32 computes a 16x128 tile of every MLP layer. Weight (B) operands
// are staged once per 128-thread block into LDS and shared by its 4 waves.
// ---------------------------------------------------------------------------

#define HID    128
#define NNODE  10000
#define NEDGE  320000
#define LAYERS 4
#define ROWPAD 136        // LDS A-row stride in bf16 elems (16B pad -> bank spread)
#define NTILES (NNODE/16) // 625
#define ETILES (NEDGE/16) // 20000
#define WSLOTS 2048       // U4s per packed 128x128 bf16 matrix (32KB)

typedef __attribute__((ext_vector_type(16))) __bf16 v16bf;
typedef __attribute__((ext_vector_type(8)))  float  v8f;

struct __attribute__((aligned(16))) U4 { unsigned int a, b, c, d; };
union Vop { v16bf v; U4 u[2]; };

// float -> bf16 (round-to-nearest-even), stored as raw u16
__device__ __forceinline__ unsigned short f2bf(float f) {
    unsigned int x = __float_as_uint(f);
    x += 0x7fffu + ((x >> 16) & 1u);
    return (unsigned short)(x >> 16);
}

__device__ __forceinline__ float silu(float x) {
    return x / (1.0f + __expf(-x));
}

__device__ __forceinline__ v8f wmma_bf16(v16bf a, v16bf b, v8f c) {
    return __builtin_amdgcn_wmma_f32_16x16x32_bf16(false, a, false, b, (short)0, c, false, false);
}

// A operand (16x32 bf16 tile) from a per-lane global row (gather).
// Lane l: m = l&15, k-half = l>>4; u[0] -> K=koff..koff+7, u[1] -> K=16+koff..
__device__ __forceinline__ v16bf load_a_global(const unsigned short* __restrict__ hb,
                                               int node, int kb, int lane) {
    const unsigned short* base = hb + node * HID + kb * 32 + ((lane >> 4) * 8);
    Vop t;
    t.u[0] = *(const U4*)(base);
    t.u[1] = *(const U4*)(base + 16);
    return t.v;
}

// A operand from LDS staging rows [16][ROWPAD]
__device__ __forceinline__ v16bf load_a_lds(const unsigned short* __restrict__ rows,
                                            int kb, int lane) {
    const unsigned short* base = rows + (lane & 15) * ROWPAD + kb * 32 + ((lane >> 4) * 8);
    Vop t;
    t.u[0] = *(const U4*)(base);
    t.u[1] = *(const U4*)(base + 16);
    return t.v;
}

// B operand (32x16 bf16 tile) from LDS-staged weights, layout [kb][nt][j][lane]
__device__ __forceinline__ v16bf load_b_lds(const U4* __restrict__ wb, int kb, int nt, int lane) {
    const U4* p = wb + ((kb * 8 + nt) * 2) * 32 + lane;
    Vop t;
    t.u[0] = p[0];
    t.u[1] = p[32];
    return t.v;
}

__device__ __forceinline__ void zero_acc(v8f acc[8]) {
    v8f z = {0.f, 0.f, 0.f, 0.f, 0.f, 0.f, 0.f, 0.f};
#pragma unroll
    for (int i = 0; i < 8; ++i) acc[i] = z;
}

// Block-cooperative stage of one packed 128x128 bf16 matrix (32KB) into LDS.
// Identity coalesced copy; prefetch a slice of the next matrix into L2.
__device__ __forceinline__ void stage_w(U4* __restrict__ dst, const U4* __restrict__ src,
                                        const U4* __restrict__ next, int tid) {
    if (next) __builtin_prefetch((const char*)next + tid * 64, 0, 1);
#pragma unroll
    for (int i = 0; i < WSLOTS / 128; ++i)
        dst[tid + i * 128] = src[tid + i * 128];
}

// acc(16x128) += A(16x128, per-lane global row `node`) x W(128x128, LDS)
__device__ __forceinline__ void gemm_aG(v8f acc[8], const unsigned short* __restrict__ hb,
                                        int node, const U4* __restrict__ wb, int lane) {
#pragma unroll
    for (int kb = 0; kb < 4; ++kb) {
        v16bf a = load_a_global(hb, node, kb, lane);
#pragma unroll
        for (int nt = 0; nt < 8; ++nt)
            acc[nt] = wmma_bf16(a, load_b_lds(wb, kb, nt, lane), acc[nt]);
    }
}

// acc(16x128) += A(16x128, LDS rows) x W(128x128, LDS)
__device__ __forceinline__ void gemm_aL(v8f acc[8], const unsigned short* __restrict__ rows,
                                        const U4* __restrict__ wb, int lane) {
#pragma unroll
    for (int kb = 0; kb < 4; ++kb) {
        v16bf a = load_a_lds(rows, kb, lane);
#pragma unroll
        for (int nt = 0; nt < 8; ++nt)
            acc[nt] = wmma_bf16(a, load_b_lds(wb, kb, nt, lane), acc[nt]);
    }
}

// ---------------------------------------------------------------------------
// Weight pack: f32 KxN (128x128 row-major) -> bf16 WMMA-B tiles.
// dword d: w=d&3, lane=(d>>2)&31, j=(d>>7)&1, nt=(d>>8)&7, kb=d>>11
// vgpr v = j*4+w ; n = nt*16+(lane&15) ; k0 = kb*32+j*16+(lane>>4)*8+2w
// ---------------------------------------------------------------------------
__global__ void pack_w_kernel(const float* __restrict__ src, uint32_t* __restrict__ dst) {
    int d = blockIdx.x * blockDim.x + threadIdx.x;
    if (d >= WSLOTS * 4) return;
    int w    = d & 3;
    int lane = (d >> 2) & 31;
    int j    = (d >> 7) & 1;
    int nt   = (d >> 8) & 7;
    int kb   = d >> 11;
    int v    = j * 4 + w;
    int n    = nt * 16 + (lane & 15);
    int k0   = kb * 32 + j * 16 + (lane >> 4) * 8 + 2 * w;
    uint32_t lo = f2bf(src[k0 * HID + n]);
    uint32_t hi = f2bf(src[(k0 + 1) * HID + n]);
    dst[d] = lo | (hi << 16);
}

// h = h_in @ emb_w + emb_b  (IN_NODE = 1), stored bf16
__global__ void embed_kernel(const float* __restrict__ h_in, const float* __restrict__ emb_w,
                             const float* __restrict__ emb_b, unsigned short* __restrict__ hbf) {
    int idx = blockIdx.x * blockDim.x + threadIdx.x;
    if (idx >= NNODE * HID) return;
    int i = idx >> 7, n = idx & 127;
    hbf[idx] = f2bf(h_in[i] * emb_w[n] + emb_b[n]);
}

// ---------------------------------------------------------------------------
// Edge kernel: 4 waves/block, wave per 16 edges. Edge MLP (2 GEMMs) ->
// coord MLP (1 GEMM + dot) -> atomic aggregation. B staged in shared LDS.
// ---------------------------------------------------------------------------
__global__ void __launch_bounds__(128) edge_kernel(
    const unsigned short* __restrict__ hbf, const float* __restrict__ xbuf,
    const float* __restrict__ eattr, const int* __restrict__ edges,
    const float* __restrict__ e_w1, const float* __restrict__ e_b1,
    const float* __restrict__ e_b2, const float* __restrict__ c_b1,
    const float* __restrict__ c_w2,
    const U4* __restrict__ pW1a, const U4* __restrict__ pW1b,
    const U4* __restrict__ pW2, const U4* __restrict__ pC1,
    float* __restrict__ m_agg, float* __restrict__ agg, float* __restrict__ cnt) {

    __shared__ U4 s_bw[WSLOTS];                      // 32KB staged weights
    __shared__ unsigned short s_rows[4][16 * ROWPAD];
    __shared__ int   s_rid[4][16];
    __shared__ float s_rad[4][16], s_e0[4][16], s_e1[4][16], s_cs[4][16];

    const int tid  = threadIdx.x;
    const int wave = tid >> 5;
    const int lane = tid & 31;
    const int tile = blockIdx.x * 4 + wave;
    const int hi   = lane >> 4;
    const int nlow = lane & 15;
    unsigned short* rows = s_rows[wave];

    float dx = 0.f, dy = 0.f, dz = 0.f;
    int r_node = 0;
    if (lane < 16) {
        int e = tile * 16 + lane;
        r_node = edges[e];
        int c  = edges[NEDGE + e];
        dx = xbuf[r_node * 3 + 0] - xbuf[c * 3 + 0];
        dy = xbuf[r_node * 3 + 1] - xbuf[c * 3 + 1];
        dz = xbuf[r_node * 3 + 2] - xbuf[c * 3 + 2];
        s_rid[wave][lane] = r_node;
        s_rad[wave][lane] = dx * dx + dy * dy + dz * dz;
        s_e0[wave][lane]  = eattr[e * 2 + 0];
        s_e1[wave][lane]  = eattr[e * 2 + 1];
        s_cs[wave][lane]  = __int_as_float(c);       // stash col id (reused later)
    }
    stage_w(s_bw, pW1a, pW1b, tid);
    __syncthreads();

    const int rid = s_rid[wave][nlow];
    const int cid = __float_as_int(s_cs[wave][nlow]);

    // per-lane per-vgpr row metadata (C layout: m = r + 8*hi)
    float rad_m[8], e0_m[8], e1_m[8];
    int   rid_m[8];
#pragma unroll
    for (int r = 0; r < 8; ++r) {
        int m    = r + 8 * hi;
        rad_m[r] = s_rad[wave][m];
        e0_m[r]  = s_e0[wave][m];
        e1_m[r]  = s_e1[wave][m];
        rid_m[r] = s_rid[wave][m];
    }

    v8f acc[8];

    // ---- GEMM1: m1 = h[row] @ W1a + h[col] @ W1b  (+ radial/eattr/bias rows)
    zero_acc(acc);
    gemm_aG(acc, hbf, rid, s_bw, lane);
    __syncthreads();
    stage_w(s_bw, pW1b, pW2, tid);
    __syncthreads();
    gemm_aG(acc, hbf, cid, s_bw, lane);
#pragma unroll
    for (int nt = 0; nt < 8; ++nt) {
        int n    = nt * 16 + nlow;
        float wr = e_w1[256 * HID + n];
        float wa = e_w1[257 * HID + n];
        float wb = e_w1[258 * HID + n];
        float b1 = e_b1[n];
#pragma unroll
        for (int r = 0; r < 8; ++r) {
            float v = acc[nt][r] + b1 + rad_m[r] * wr + e0_m[r] * wa + e1_m[r] * wb;
            rows[(r + 8 * hi) * ROWPAD + n] = f2bf(silu(v));
        }
    }
    __syncthreads();
    stage_w(s_bw, pW2, pC1, tid);
    __syncthreads();

    // ---- GEMM2: m = silu(m1 @ e_w2 + b2); aggregate to m_agg[row]
    zero_acc(acc);
    gemm_aL(acc, rows, s_bw, lane);
    __syncthreads();
    stage_w(s_bw, pC1, (const U4*)0, tid);
#pragma unroll
    for (int nt = 0; nt < 8; ++nt) {
        int n    = nt * 16 + nlow;
        float b2 = e_b2[n];
#pragma unroll
        for (int r = 0; r < 8; ++r) {
            float m = silu(acc[nt][r] + b2);
            atomicAdd(&m_agg[rid_m[r] * HID + n], m);
            rows[(r + 8 * hi) * ROWPAD + n] = f2bf(m);
        }
    }
    __syncthreads();

    // ---- GEMM3: c1 = silu(m @ c_w1 + c_b1); cscal = c1 . c_w2
    zero_acc(acc);
    gemm_aL(acc, rows, s_bw, lane);
    float part[8];
#pragma unroll
    for (int r = 0; r < 8; ++r) part[r] = 0.f;
#pragma unroll
    for (int nt = 0; nt < 8; ++nt) {
        int n    = nt * 16 + nlow;
        float b  = c_b1[n];
        float w2 = c_w2[n];
#pragma unroll
        for (int r = 0; r < 8; ++r)
            part[r] += silu(acc[nt][r] + b) * w2;
    }
#pragma unroll
    for (int r = 0; r < 8; ++r) {
        float p = part[r];
        p += __shfl_xor(p, 1, 32);
        p += __shfl_xor(p, 2, 32);
        p += __shfl_xor(p, 4, 32);
        p += __shfl_xor(p, 8, 32);
        if (nlow == 0) s_cs[wave][r + 8 * hi] = p;
    }
    __syncthreads();

    if (lane < 16) {
        float cs = s_cs[wave][lane];
        atomicAdd(&agg[0 * NNODE + r_node], dx * cs);
        atomicAdd(&agg[1 * NNODE + r_node], dy * cs);
        atomicAdd(&agg[2 * NNODE + r_node], dz * cs);
        atomicAdd(&cnt[r_node], 1.0f);
    }
}

// ---------------------------------------------------------------------------
// Node kernel: 4 waves/block, wave per 16 nodes. Velocity MLP, x update,
// node MLP, h update (bf16, in place — each row owned by exactly one wave).
// Tail waves masked via `valid`; barriers stay block-uniform.
// ---------------------------------------------------------------------------
__global__ void __launch_bounds__(128) node_kernel(
    unsigned short* __restrict__ hbf, float* __restrict__ xbuf,
    const float* __restrict__ vel,
    const float* __restrict__ m_agg, const float* __restrict__ agg,
    const float* __restrict__ cnt,
    const float* __restrict__ v_b1, const float* __restrict__ v_w2,
    const float* __restrict__ v_b2,
    const float* __restrict__ n_b1, const float* __restrict__ n_b2,
    const U4* __restrict__ pV1, const U4* __restrict__ pNa,
    const U4* __restrict__ pNb, const U4* __restrict__ pN2) {

    __shared__ U4 s_bw[WSLOTS];
    __shared__ unsigned short s_rows[4][16 * ROWPAD];
    __shared__ float s_vs[4][16];

    const int tid  = threadIdx.x;
    const int wave = tid >> 5;
    const int lane = tid & 31;
    const int tile = blockIdx.x * 4 + wave;
    const bool valid = (tile < NTILES);
    const int base = (valid ? tile : 0) * 16;
    const int hi   = lane >> 4;
    const int nlow = lane & 15;
    unsigned short* rows = s_rows[wave];

    v8f acc[8];

    // ---- velocity path: v1 = silu(h @ v_w1 + b1); vscal = v1 . v_w2 + v_b2
    stage_w(s_bw, pV1, pNa, tid);
    __syncthreads();
    zero_acc(acc);
    gemm_aG(acc, hbf, base + nlow, s_bw, lane);
    float part[8];
#pragma unroll
    for (int r = 0; r < 8; ++r) part[r] = 0.f;
#pragma unroll
    for (int nt = 0; nt < 8; ++nt) {
        int n    = nt * 16 + nlow;
        float b  = v_b1[n];
        float w2 = v_w2[n];
#pragma unroll
        for (int r = 0; r < 8; ++r)
            part[r] += silu(acc[nt][r] + b) * w2;
    }
#pragma unroll
    for (int r = 0; r < 8; ++r) {
        float p = part[r];
        p += __shfl_xor(p, 1, 32);
        p += __shfl_xor(p, 2, 32);
        p += __shfl_xor(p, 4, 32);
        p += __shfl_xor(p, 8, 32);
        if (nlow == 0) s_vs[wave][r + 8 * hi] = p;
    }
    __syncthreads();

    // ---- x update: x += agg/clip(cnt,1) + vscal * vel
    if (valid && lane < 16) {
        int i = base + lane;
        float cn = fmaxf(cnt[i], 1.0f);
        float vs = s_vs[wave][lane] + v_b2[0];
#pragma unroll
        for (int d = 0; d < 3; ++d) {
            float xv = xbuf[i * 3 + d];
            xv += agg[d * NNODE + i] / cn;
            xv += vs * vel[i * 3 + d];
            xbuf[i * 3 + d] = xv;
        }
    }

    // ---- stage m_agg tile as bf16 A rows
    for (int t = lane; t < 16 * HID; t += 32) {
        int m = t >> 7, k = t & 127;
        rows[m * ROWPAD + k] = f2bf(m_agg[(base + m) * HID + k]);
    }
    __syncthreads();
    stage_w(s_bw, pNa, pNb, tid);
    __syncthreads();

    // ---- node MLP: n1 = silu(h @ Na + m_agg @ Nb + b1)
    zero_acc(acc);
    gemm_aG(acc, hbf, base + nlow, s_bw, lane);
    __syncthreads();
    stage_w(s_bw, pNb, pN2, tid);
    __syncthreads();
    gemm_aL(acc, rows, s_bw, lane);
    __syncthreads();
    stage_w(s_bw, pN2, (const U4*)0, tid);
#pragma unroll
    for (int nt = 0; nt < 8; ++nt) {
        int n   = nt * 16 + nlow;
        float b = n_b1[n];
#pragma unroll
        for (int r = 0; r < 8; ++r)
            rows[(r + 8 * hi) * ROWPAD + n] = f2bf(silu(acc[nt][r] + b));
    }
    __syncthreads();

    // ---- h_new = n1 @ n_w2 + b2 (no activation), write bf16 h in place
    zero_acc(acc);
    gemm_aL(acc, rows, s_bw, lane);
    if (valid) {
#pragma unroll
        for (int nt = 0; nt < 8; ++nt) {
            int n   = nt * 16 + nlow;
            float b = n_b2[n];
#pragma unroll
            for (int r = 0; r < 8; ++r)
                hbf[(base + r + 8 * hi) * HID + n] = f2bf(acc[nt][r] + b);
        }
    }
}

// ---------------------------------------------------------------------------
extern "C" void kernel_launch(void* const* d_in, const int* in_sizes, int n_in,
                              void* d_out, int out_size, void* d_ws, size_t ws_size,
                              hipStream_t stream) {
    const float* h_in  = (const float*)d_in[0];
    const float* x_in  = (const float*)d_in[1];
    const float* vel   = (const float*)d_in[2];
    const float* eattr = (const float*)d_in[3];
    const int*   edges = (const int*)d_in[4];
    const float* emb_w = (const float*)d_in[5];
    const float* emb_b = (const float*)d_in[6];
    auto LP = [&](int l, int k) { return (const float*)d_in[7 + l * 15 + k]; };
    // layer param order: e_w1, e_b1, e_w2, e_b2, c_w1, c_b1, c_w2,
    //                    v_w1, v_b1, v_w2, v_b2, n_w1, n_b1, n_w2, n_b2

    char* ws = (char*)d_ws;
    size_t off = 0;
    auto carve = [&](size_t bytes) {
        size_t o = off;
        off = (off + bytes + 255) & ~(size_t)255;
        return o;
    };
    float*          xbuf   = (float*)(ws + carve((size_t)NNODE * 3 * 4));
    unsigned short* hbf    = (unsigned short*)(ws + carve((size_t)NNODE * HID * 2));
    float*          m_agg  = (float*)(ws + carve((size_t)NNODE * HID * 4));
    float*          agg    = (float*)(ws + carve((size_t)NNODE * 3 * 4));
    float*          cnt    = (float*)(ws + carve((size_t)NNODE * 4));
    char*           packed = ws + carve((size_t)LAYERS * 8 * HID * HID * 2);
    auto slot = [&](int l, int j) { return (U4*)(packed + (size_t)(l * 8 + j) * (HID * HID * 2)); };

    // ---- pack weights to bf16 WMMA-B layout (8 matrices per layer)
    for (int l = 0; l < LAYERS; ++l) {
        const float* ew1 = LP(l, 0);
        pack_w_kernel<<<32, 256, 0, stream>>>(ew1,                 (uint32_t*)slot(l, 0)); // W1 h[row] part
        pack_w_kernel<<<32, 256, 0, stream>>>(ew1 + 128 * HID,     (uint32_t*)slot(l, 1)); // W1 h[col] part
        pack_w_kernel<<<32, 256, 0, stream>>>(LP(l, 2),            (uint32_t*)slot(l, 2)); // e_w2
        pack_w_kernel<<<32, 256, 0, stream>>>(LP(l, 4),            (uint32_t*)slot(l, 3)); // c_w1
        pack_w_kernel<<<32, 256, 0, stream>>>(LP(l, 7),            (uint32_t*)slot(l, 4)); // v_w1
        pack_w_kernel<<<32, 256, 0, stream>>>(LP(l, 11),           (uint32_t*)slot(l, 5)); // n_w1 h part
        pack_w_kernel<<<32, 256, 0, stream>>>(LP(l, 11) + 128*HID, (uint32_t*)slot(l, 6)); // n_w1 m part
        pack_w_kernel<<<32, 256, 0, stream>>>(LP(l, 13),           (uint32_t*)slot(l, 7)); // n_w2
    }

    // ---- embedding + x init
    embed_kernel<<<(NNODE * HID + 255) / 256, 256, 0, stream>>>(h_in, emb_w, emb_b, hbf);
    hipMemcpyAsync(xbuf, x_in, (size_t)NNODE * 3 * 4, hipMemcpyDeviceToDevice, stream);

    // ---- layers
    for (int l = 0; l < LAYERS; ++l) {
        hipMemsetAsync(m_agg, 0, (size_t)NNODE * HID * 4, stream);
        hipMemsetAsync(agg,   0, (size_t)NNODE * 3 * 4, stream);
        hipMemsetAsync(cnt,   0, (size_t)NNODE * 4, stream);

        edge_kernel<<<ETILES / 4, 128, 0, stream>>>(
            hbf, xbuf, eattr, edges,
            LP(l, 0), LP(l, 1), LP(l, 3), LP(l, 5), LP(l, 6),
            slot(l, 0), slot(l, 1), slot(l, 2), slot(l, 3),
            m_agg, agg, cnt);

        node_kernel<<<(NTILES + 3) / 4, 128, 0, stream>>>(
            hbf, xbuf, vel, m_agg, agg, cnt,
            LP(l, 8), LP(l, 9), LP(l, 10), LP(l, 12), LP(l, 14),
            slot(l, 4), slot(l, 5), slot(l, 6), slot(l, 7));
    }

    // ---- output = final x
    hipMemcpyAsync(d_out, xbuf, (size_t)NNODE * 3 * 4, hipMemcpyDeviceToDevice, stream);
}
